// ExLRestSelfAtten_42236708389609
// MI455X (gfx1250) — matrix operand order repaired and checked by reading, not compiled, not run on hardware
//
#include <hip/hip_runtime.h>
#include <hip/hip_bf16.h>
#include <stdint.h>

typedef __attribute__((ext_vector_type(16))) _Float16 v16h;
typedef __attribute__((ext_vector_type(8)))  _Float16 v8h;
typedef __attribute__((ext_vector_type(8)))  float    v8f;

union FragA { v16h v; uint32_t u[8]; _Float16 h[16]; };
union FragC { v8f  v; float    f[8]; };

// A-matrix (16x32 f16) K index of 32-bit pair p for lane-half kh
// lanes 0-15: K {0..7,16..23}; lanes 16-31: K {8..15,24..31}
__device__ __forceinline__ int a_pair_k(int p, int kh) {
    return ((p & 4) ? 16 : 0) + 8 * kh + 2 * (p & 3);
}
// B-matrix (32x16 f16) K index of pair p: lanes 0-15 K {0..7,16..23}; lanes 16-31 K {8..15,24..31}
__device__ __forceinline__ int b_pair_k(int p, int kh) {
    return ((p & 4) ? 16 : 0) + 8 * kh + 2 * (p & 3);
}

// CDNA5 async global->LDS copy, 16B per lane, tracked by ASYNCcnt
__device__ __forceinline__ void async_copy_b128(uint32_t lds_off, const void* gaddr) {
    asm volatile("global_load_async_to_lds_b128 %0, %1, off"
                 :: "v"(lds_off), "v"(gaddr) : "memory");
}
__device__ __forceinline__ void wait_async0() {
    asm volatile("s_wait_asynccnt 0x0" ::: "memory");
}

#define BM 128
#define BN 64
#define BK 32

// ---------------- fp32 -> f16 bulk convert (packed, b64 stores) ----------------
__global__ __launch_bounds__(256) void cvt_f32_f16(
    const float* __restrict__ s, _Float16* __restrict__ d, int n)
{
    int i = (blockIdx.x * 256 + threadIdx.x) * 4;
    if (i < n) {
        float4 v = *(const float4*)(s + i);
        union { _Float16 h[4]; uint64_t u; } p;
        p.h[0] = (_Float16)v.x; p.h[1] = (_Float16)v.y;
        p.h[2] = (_Float16)v.z; p.h[3] = (_Float16)v.w;
        *(uint64_t*)(d + i) = p.u;
    }
}

// ---------------- unified f16 GEMM: O = [relu](A @ B + bias) ----------------
// A: M x K f16 row-major; B: K x N f16 row-major; O: M x N f16
__global__ __launch_bounds__(256) void gemm_f16(
    const _Float16* __restrict__ A, const _Float16* __restrict__ Bw,
    const float* __restrict__ bias, _Float16* __restrict__ O,
    int M, int K, int N, int relu)
{
    __shared__ _Float16 sA[BM * BK];   // [r][k]
    __shared__ _Float16 sB[BN * BK];   // [n][k] (transposed during staging)
    const int tid = threadIdx.x, lane = tid & 31, wv = tid >> 5;
    const int kh = lane >> 4, l16 = lane & 15;
    const int m0 = blockIdx.x * BM, n0 = blockIdx.y * BN;
    const uint32_t ldsA = (uint32_t)(uintptr_t)sA;   // low 32 bits = LDS offset
    FragC acc[4] = {};

    for (int kb = 0; kb < K; kb += BK) {
        // A tile: 128x32 f16 = 8KB, async DMA to LDS, 2 x 16B chunks per thread
        #pragma unroll
        for (int cc = 0; cc < 2; ++cc) {
            int c = tid + cc * 256;                   // 0..511
            int r = c >> 2, ko = (c & 3) * 8;
            async_copy_b128(ldsA + (uint32_t)c * 16,
                            A + (size_t)(m0 + r) * K + kb + ko);
        }
        // B tile: coalesced 16B f16 load, transpose-scatter to LDS [n][k]
        {
            int k = tid >> 3, n8 = (tid & 7) * 8;
            v8h bv = *(const v8h*)(Bw + (size_t)(kb + k) * N + n0 + n8);
            #pragma unroll
            for (int j = 0; j < 8; ++j) sB[(n8 + j) * BK + k] = bv[j];
        }
        // prefetch next k tiles into L2
        if (kb + BK < K) {
            __builtin_prefetch(A + (size_t)(m0 + (tid >> 2)) * K + kb + BK, 0, 0);
            __builtin_prefetch(Bw + (size_t)(kb + BK + (tid >> 3)) * N + n0, 0, 0);
        }
        wait_async0();
        __syncthreads();

        const uint32_t* sA32 = (const uint32_t*)sA;
        const uint32_t* sB32 = (const uint32_t*)sB;
        FragA a;
        {
            int row = wv * 16 + l16;
            #pragma unroll
            for (int p = 0; p < 8; ++p)
                a.u[p] = sA32[(row * BK + a_pair_k(p, kh)) >> 1];
        }
        #pragma unroll
        for (int t = 0; t < 4; ++t) {
            FragA b;
            int n = t * 16 + l16;
            #pragma unroll
            for (int p = 0; p < 8; ++p)
                b.u[p] = sB32[(n * BK + b_pair_k(p, kh)) >> 1];
            acc[t].v = __builtin_amdgcn_wmma_f32_16x16x32_f16(
                false, a.v, false, b.v, (short)0, acc[t].v, false, false);
        }
        __syncthreads();
    }
    #pragma unroll
    for (int t = 0; t < 4; ++t) {
        #pragma unroll
        for (int r = 0; r < 8; ++r) {
            int m = m0 + wv * 16 + r + 8 * kh;
            int n = n0 + t * 16 + l16;
            float v = acc[t].f[r];
            if (relu) v = fmaxf(v + bias[n], 0.0f);
            O[(size_t)m * N + n] = (_Float16)v;
        }
    }
}

// ---------------- Local attention ----------------
// One block per (batch b, 16-row s-block). Window = rows s0-16 .. s0+31 (48 rows).
// scores(16x48) = Q_blk * Hwin^T ; softmax with reference zero-padding semantics;
// out(16x1024) = P(16x64 zero-padded) * Hwin.
__global__ __launch_bounds__(128) void attn_local(
    const _Float16* __restrict__ Hf,   // (B*S) x Hdim, f16
    const _Float16* __restrict__ Qf,   // (B*S) x Hdim, f16
    float* __restrict__ Out,           // (B*S) x Hdim, f32
    int S, int Hdim)
{
    __shared__ float    sScores[4][16 * 48];
    __shared__ _Float16 sP[16 * 64];
    const int tid = threadIdx.x, lane = tid & 31, wv = tid >> 5;
    const int kh = lane >> 4, l16 = lane & 15;
    const int spb = S / 16;
    const int b  = blockIdx.x / spb;
    const int s0 = (blockIdx.x % spb) * 16;
    const size_t rowbase = (size_t)b * S;

    // ---- stage 1: scores, K split across 4 waves ----
    FragC sc[3] = {};
    for (int ks = wv; ks < Hdim / 32; ks += 4) {
        const int kb = ks * 32;
        FragA a;
        {
            const size_t qoff = (rowbase + s0 + l16) * (size_t)Hdim + kb;
            #pragma unroll
            for (int p = 0; p < 8; ++p)
                a.u[p] = *(const uint32_t*)(Qf + qoff + a_pair_k(p, kh));
        }
        #pragma unroll
        for (int t = 0; t < 3; ++t) {
            FragA bf;
            int nbr = s0 - 16 + 16 * t + l16;          // window row for column l16
            int nbrc = nbr < 0 ? 0 : (nbr >= S ? S - 1 : nbr);
            const size_t hoff = (rowbase + nbrc) * (size_t)Hdim + kb;
            #pragma unroll
            for (int p = 0; p < 8; ++p)
                bf.u[p] = *(const uint32_t*)(Hf + hoff + b_pair_k(p, kh));
            sc[t].v = __builtin_amdgcn_wmma_f32_16x16x32_f16(
                false, a.v, false, bf.v, (short)0, sc[t].v, false, false);
        }
    }
    #pragma unroll
    for (int t = 0; t < 3; ++t)
        #pragma unroll
        for (int r = 0; r < 8; ++r)
            sScores[wv][(r + 8 * kh) * 48 + 16 * t + l16] = sc[t].f[r];
    __syncthreads();

    // ---- softmax (one lane per s-row); reference semantics:
    // out-of-window -> excluded; in-window but out-of-range -> score 0 kept in
    // denominator, output weight forced to 0 (zero-padded neighbor). ----
    if (tid < 16) {
        const int r = tid;
        float mx = -1e30f;
        for (int j = 0; j < 48; ++j) {
            int delta = j - 16 - r;
            int nbr = s0 - 16 + j;
            bool inwin = (delta >= -16) && (delta <= 16);
            bool inrng = (nbr >= 0) && (nbr < S);
            float v = (sScores[0][r*48+j] + sScores[1][r*48+j] +
                       sScores[2][r*48+j] + sScores[3][r*48+j]) * 0.03125f;
            v = inwin ? (inrng ? v : 0.0f) : -1e30f;
            sScores[0][r*48+j] = v;
            mx = fmaxf(mx, v);
        }
        float sum = 0.0f;
        for (int j = 0; j < 48; ++j) {
            int delta = j - 16 - r;
            bool inwin = (delta >= -16) && (delta <= 16);
            float ev = inwin ? __expf(sScores[0][r*48+j] - mx) : 0.0f;
            sScores[0][r*48+j] = ev;
            sum += ev;
        }
        const float rs = 1.0f / sum;
        for (int j = 0; j < 64; ++j) {
            float w = 0.0f;
            if (j < 48) {
                int delta = j - 16 - r;
                int nbr = s0 - 16 + j;
                bool keep = (delta >= -16) && (delta <= 16) && (nbr >= 0) && (nbr < S);
                w = keep ? sScores[0][r*48+j] * rs : 0.0f;
            }
            sP[r * 64 + j] = (_Float16)w;
        }
    }
    __syncthreads();

    // ---- stage 2: out = P(16x64) * Hwin(64x16-tiles), K padded to 64 ----
    const uint32_t* sP32 = (const uint32_t*)sP;
    for (int ct = wv; ct < Hdim / 16; ct += 4) {
        const int n0c = ct * 16;
        FragC o = {};
        #pragma unroll
        for (int ks = 0; ks < 2; ++ks) {
            const int kb = ks * 32;
            FragA a;
            #pragma unroll
            for (int p = 0; p < 8; ++p)
                a.u[p] = sP32[(l16 * 64 + kb + a_pair_k(p, kh)) >> 1];
            FragA bf;
            #pragma unroll
            for (int e = 0; e < 16; ++e) {
                int k = b_pair_k(e >> 1, kh) + (e & 1);
                int nbr = s0 - 16 + kb + k;
                int nbrc = nbr < 0 ? 0 : (nbr >= S ? S - 1 : nbr);  // P==0 there
                bf.h[e] = Hf[(rowbase + nbrc) * (size_t)Hdim + n0c + l16];
            }
            o.v = __builtin_amdgcn_wmma_f32_16x16x32_f16(
                false, a.v, false, bf.v, (short)0, o.v, false, false);
        }
        #pragma unroll
        for (int r = 0; r < 8; ++r)
            Out[(rowbase + s0 + r + 8 * kh) * (size_t)Hdim + n0c + l16] = o.f[r];
    }
}

extern "C" void kernel_launch(void* const* d_in, const int* in_sizes, int n_in,
                              void* d_out, int out_size, void* d_ws, size_t ws_size,
                              hipStream_t stream) {
    const float* x  = (const float*)d_in[0];
    const float* W1 = (const float*)d_in[1];
    const float* b1 = (const float*)d_in[2];
    const float* Wq = (const float*)d_in[3];
    // atten_size = 16 (hardcoded, matches reference setup)
    const int Bb = 4, S = 2048, IN = 512, Hh = 1024;
    const int M = Bb * S;

    _Float16* xf   = (_Float16*)d_ws;                 // M x IN      (8 MB)
    _Float16* w1f  = xf  + (size_t)M * IN;            // IN x Hh     (1 MB)
    _Float16* wqf  = w1f + (size_t)IN * Hh;           // Hh x Hh     (2 MB)
    _Float16* hbuf = wqf + (size_t)Hh * Hh;           // M x Hh      (16 MB)
    _Float16* qbuf = hbuf + (size_t)M * Hh;           // M x Hh      (16 MB)
    float* out = (float*)d_out;

    dim3 blk(256);
    cvt_f32_f16<<<dim3((M * IN) / 4 / 256), blk, 0, stream>>>(x,  xf,  M * IN);
    cvt_f32_f16<<<dim3((IN * Hh) / 4 / 256), blk, 0, stream>>>(W1, w1f, IN * Hh);
    cvt_f32_f16<<<dim3((Hh * Hh) / 4 / 256), blk, 0, stream>>>(Wq, wqf, Hh * Hh);

    dim3 g1(M / BM, Hh / BN);
    gemm_f16<<<g1, blk, 0, stream>>>(xf,   w1f, b1,      hbuf, M, IN, Hh, 1);
    gemm_f16<<<g1, blk, 0, stream>>>(hbuf, wqf, nullptr, qbuf, M, Hh, Hh, 0);
    attn_local<<<dim3(M / 16), dim3(128), 0, stream>>>(hbuf, qbuf, out, S, Hh);
}
